// HeteroGNN_22436909154370
// MI455X (gfx1250) — matrix operand order repaired and checked by reading, not compiled, not run on hardware
//
#include <hip/hip_runtime.h>

typedef __attribute__((ext_vector_type(16))) _Float16 v16h;
typedef __attribute__((ext_vector_type(8)))  float    v8f;

#define TPB 256

// ---------------- utility kernels ----------------

__global__ __launch_bounds__(TPB) void fill_f32(float* p, int n, float v) {
    int i = blockIdx.x * TPB + threadIdx.x;
    if (i < n) p[i] = v;
}

// prod[p,j] = b_gcn[j] + b_gat[j] + b_to[j] + b_from[j] + b_div[j]
__global__ __launch_bounds__(TPB) void bias_init(float* __restrict__ prod,
        const float* __restrict__ b0, const float* __restrict__ b1,
        const float* __restrict__ b2, const float* __restrict__ b3,
        const float* __restrict__ b4, int n) {
    int i = blockIdx.x * TPB + threadIdx.x;
    if (i >= n) return;
    int j = i & 127;
    prod[i] = b0[j] + b1[j] + b2[j] + b3[j] + b4[j];
}

__global__ __launch_bounds__(TPB) void add3(const float* __restrict__ a,
        const float* __restrict__ b, const float* __restrict__ c,
        float* __restrict__ o, int n) {
    int i = blockIdx.x * TPB + threadIdx.x;
    if (i < n) o[i] = a[i] + b[i] + c[i];
}

__global__ __launch_bounds__(TPB) void degree_count(const int* __restrict__ src,
        const int* __restrict__ dst, float* ds, float* dd, int E) {
    int e = blockIdx.x * TPB + threadIdx.x;
    if (e >= E) return;
    atomicAdd(&ds[src[e]], 1.0f);
    atomicAdd(&dd[dst[e]], 1.0f);
}

// out[k] = dot(W[k,0:128], a[0:128])   (W is 128x128 row-major)
__global__ __launch_bounds__(TPB) void matvec128(const float* __restrict__ W,
        const float* __restrict__ a, float* __restrict__ out) {
    int k = blockIdx.x * TPB + threadIdx.x;
    if (k >= 128) return;
    float s = 0.f;
    #pragma unroll 8
    for (int j = 0; j < 128; ++j) s += W[k * 128 + j] * a[j];
    out[k] = s;
}

// out[row] = dot(X[row,0:128], v)  — one wave per row, float4 per lane
__global__ __launch_bounds__(TPB) void rowdot128(const float* __restrict__ X,
        const float* __restrict__ v, float* __restrict__ out, int M) {
    int idx = blockIdx.x * TPB + threadIdx.x;
    int row = idx >> 5, lane = idx & 31;
    if (row >= M) return;
    float4 x = ((const float4*)(X + (long)row * 128))[lane];
    float4 w = ((const float4*)v)[lane];
    float s = x.x * w.x + x.y * w.y + x.z * w.z + x.w * w.w;
    #pragma unroll
    for (int off = 16; off > 0; off >>= 1) s += __shfl_down(s, off, 32);
    if (lane == 0) out[row] = s;
}

// ---------------- GAT softmax pieces ----------------

__device__ __forceinline__ unsigned enc_f32(float f) {
    unsigned b = __float_as_uint(f);
    return (b & 0x80000000u) ? ~b : (b | 0x80000000u);
}
__device__ __forceinline__ float dec_f32(unsigned e) {
    return (e & 0x80000000u) ? __uint_as_float(e ^ 0x80000000u)
                             : __uint_as_float(~e);
}

__global__ __launch_bounds__(TPB) void gat_logit_max(const int* __restrict__ src,
        const int* __restrict__ dst, const float* __restrict__ ls,
        const float* __restrict__ ld, float* __restrict__ logit,
        unsigned* __restrict__ mbits, int E) {
    int e = blockIdx.x * TPB + threadIdx.x;
    if (e >= E) return;
    float l = ls[src[e]] + ld[dst[e]];
    l = (l > 0.f) ? l : 0.2f * l;             // leaky_relu(., 0.2)
    logit[e] = l;
    atomicMax(&mbits[dst[e]], enc_f32(l));
}

__global__ __launch_bounds__(TPB) void gat_exp_den(const int* __restrict__ dst,
        float* __restrict__ logit, const unsigned* __restrict__ mbits,
        float* __restrict__ den, int E) {
    int e = blockIdx.x * TPB + threadIdx.x;
    if (e >= E) return;
    float m = dec_f32(mbits[dst[e]]);
    float ex = expf(logit[e] - m);
    logit[e] = ex;                             // reuse as numerator
    atomicAdd(&den[dst[e]], ex);
}

// ---------------- edge scatters (32 lanes/edge, float4 per lane) ----------------

__global__ __launch_bounds__(TPB) void scatter_gcn(const float* __restrict__ h,
        const int* __restrict__ src, const int* __restrict__ dst,
        const float* __restrict__ ds, const float* __restrict__ dd,
        float* __restrict__ out, int E) {
    int idx = blockIdx.x * TPB + threadIdx.x;
    int e = idx >> 5;
    if (e >= E) return;
    int lane = idx & 31;
    int s = src[e], d = dst[e];
    float norm = rsqrtf(ds[s] * dd[d]);
    float4 v = ((const float4*)(h + (long)s * 128))[lane];
    float* o = out + (long)d * 128 + lane * 4;
    atomicAdd(o + 0, norm * v.x); atomicAdd(o + 1, norm * v.y);
    atomicAdd(o + 2, norm * v.z); atomicAdd(o + 3, norm * v.w);
}

__global__ __launch_bounds__(TPB) void scatter_gat(const float* __restrict__ hs,
        const int* __restrict__ src, const int* __restrict__ dst,
        const float* __restrict__ num, const float* __restrict__ den,
        float* __restrict__ out, int E) {
    int idx = blockIdx.x * TPB + threadIdx.x;
    int e = idx >> 5;
    if (e >= E) return;
    int lane = idx & 31;
    int s = src[e], d = dst[e];
    float alpha = num[e] / den[d];
    float4 v = ((const float4*)(hs + (long)s * 128))[lane];
    float* o = out + (long)d * 128 + lane * 4;
    atomicAdd(o + 0, alpha * v.x); atomicAdd(o + 1, alpha * v.y);
    atomicAdd(o + 2, alpha * v.z); atomicAdd(o + 3, alpha * v.w);
}

__global__ __launch_bounds__(TPB) void scatter_sage(const float* __restrict__ x,
        const int* __restrict__ src, const int* __restrict__ dst,
        float* __restrict__ sum, float* __restrict__ cnt, int E) {
    int idx = blockIdx.x * TPB + threadIdx.x;
    int e = idx >> 5;
    if (e >= E) return;
    int lane = idx & 31;
    int s = src[e], d = dst[e];
    float4 v = ((const float4*)(x + (long)s * 128))[lane];
    float* o = sum + (long)d * 128 + lane * 4;
    atomicAdd(o + 0, v.x); atomicAdd(o + 1, v.y);
    atomicAdd(o + 2, v.z); atomicAdd(o + 3, v.w);
    if (lane == 0) atomicAdd(&cnt[d], 1.0f);
}

// ---------------- weight pre-pack into WMMA B-fragment order ----------------
// Bp[((c*TN + t)*32 + lane)*16 + j] = (f16) B[(c*32 + (lane>>4)*8 + (j&7) + (j>>3)*16)*N + t*16 + (lane&15)]
// so a wave's B fragment for K-chunk c, column-tile t is 32B contiguous per lane.
template <int TN>
__global__ __launch_bounds__(TPB) void pack_b(const float* __restrict__ B,
                                              _Float16* __restrict__ Bp) {
    const int N = TN * 16;
    int idx = blockIdx.x * TPB + threadIdx.x;
    if (idx >= 4 * TN * 32 * 16) return;
    int j    = idx & 15;
    int lane = (idx >> 4) & 31;
    int ct   = idx >> 9;               // c*TN + t
    int t = ct % TN, c = ct / TN;
    int kb  = c * 32 + (lane >> 4) * 8 + (j & 7) + (j >> 3) * 16;
    int col = t * 16 + (lane & 15);
    Bp[idx] = (_Float16)B[kb * N + col];
}

// ---------------- WMMA GEMM: C[M, TN*16] (+)= op(A[M,128]) @ B ----------------
// One wave computes a full 16-row strip: TN 16x16 tiles, K=128 as 4 WMMA steps.
// A fragment loaded/converted once per K-chunk, reused for all TN WMMAs.
// RELU_A: relu on A load.  MEAN: scale A row by 1/max(cnt[row],1) (SAGE mean).
// bias!=null: add bias[col] on store.  ACC: C += instead of C =.
template <int TN, bool RELU_A, bool ACC, bool MEAN>
__global__ __launch_bounds__(TPB) void gemm_f16p(const float* __restrict__ A,
        const _Float16* __restrict__ Bp, float* __restrict__ C, int M,
        const float* __restrict__ cnt, const float* __restrict__ bias) {
    const int K = 128;
    const int N = TN * 16;
    int wave = threadIdx.x >> 5;
    int lane = threadIdx.x & 31;
    int mt = blockIdx.x * 8 + wave;                  // wave-uniform
    if (mt >= (M >> 4)) return;                      // whole-wave exit
    int lhalf = lane >> 4;      // 0: lanes 0-15, 1: lanes 16-31
    int lmod  = lane & 15;

    int arow = mt * 16 + lmod;
    const float* Arow = A + (long)arow * K;
    float scale = 1.0f;
    if (MEAN) scale = 1.0f / fmaxf(cnt[arow], 1.0f);

    v8f acc[TN];
    #pragma unroll
    for (int t = 0; t < TN; ++t) acc[t] = (v8f){};

    #pragma unroll
    for (int c = 0; c < 4; ++c) {
        // ---- A fragment (16-bit A layout: lanes<16 hold K=kk+0..7 & kk+16..23;
        //      lanes>=16 hold K=kk+8..15 & kk+24..31) ----
        int k0 = c * 32 + lhalf * 8;
        float4 p0 = *(const float4*)(Arow + k0);
        float4 p1 = *(const float4*)(Arow + k0 + 4);
        float4 q0 = *(const float4*)(Arow + k0 + 16);
        float4 q1 = *(const float4*)(Arow + k0 + 20);
        float af[16] = {p0.x, p0.y, p0.z, p0.w, p1.x, p1.y, p1.z, p1.w,
                        q0.x, q0.y, q0.z, q0.w, q1.x, q1.y, q1.z, q1.w};
        v16h a;
        #pragma unroll
        for (int j = 0; j < 16; ++j) {
            float x = af[j];
            if (RELU_A) x = fmaxf(x, 0.f);
            if (MEAN)   x *= scale;
            a[j] = (_Float16)x;
        }
        // ---- TN pre-packed B fragments, contiguous 32B per lane ----
        #pragma unroll
        for (int t = 0; t < TN; ++t) {
            v16h b = *(const v16h*)(Bp + ((long)(c * TN + t) * 32 + lane) * 16);
            acc[t] = __builtin_amdgcn_wmma_f32_16x16x32_f16(
                        false, a, false, b, (short)0, acc[t], false, false);
        }
    }
    // ---- store: VGPR r -> row mt*16 + r + lhalf*8, col t*16 + lmod ----
    #pragma unroll
    for (int t = 0; t < TN; ++t) {
        int bcol = t * 16 + lmod;
        float badd = bias ? bias[bcol] : 0.f;
        #pragma unroll
        for (int r = 0; r < 8; ++r) {
            int row = mt * 16 + lhalf * 8 + r;
            long off = (long)row * N + bcol;
            float v = acc[t][r] + badd;
            if (ACC) C[off] += v; else C[off] = v;
        }
    }
}

// ---------------- host launcher ----------------

extern "C" void kernel_launch(void* const* d_in, const int* in_sizes, int n_in,
                              void* d_out, int out_size, void* d_ws, size_t ws_size,
                              hipStream_t stream) {
    const float* x_cust = (const float*)d_in[0];
    const float* x_prod = (const float*)d_in[1];
    const int* src_pur = (const int*)d_in[2];
    const int* dst_pur = (const int*)d_in[3];
    const int* src_red = (const int*)d_in[4];
    const int* dst_red = (const int*)d_in[5];
    const int* src_to  = (const int*)d_in[6];
    const int* dst_to  = (const int*)d_in[7];
    const int* src_fr  = (const int*)d_in[8];
    const int* dst_fr  = (const int*)d_in[9];
    const int* src_dv  = (const int*)d_in[10];
    const int* dst_dv  = (const int*)d_in[11];
    const float* W_gcn  = (const float*)d_in[12];
    const float* b_gcn  = (const float*)d_in[13];
    const float* Ws_gat = (const float*)d_in[14];
    const float* Wd_gat = (const float*)d_in[15];
    const float* a_s    = (const float*)d_in[16];
    const float* a_d    = (const float*)d_in[17];
    const float* b_gat  = (const float*)d_in[18];
    const float* Wl_to  = (const float*)d_in[19];
    const float* b_to   = (const float*)d_in[20];
    const float* Wr_to  = (const float*)d_in[21];
    const float* Wl_fr  = (const float*)d_in[22];
    const float* b_fr   = (const float*)d_in[23];
    const float* Wr_fr  = (const float*)d_in[24];
    const float* Wl_dv  = (const float*)d_in[25];
    const float* b_dv   = (const float*)d_in[26];
    const float* Wr_dv  = (const float*)d_in[27];
    const float* W_out  = (const float*)d_in[28];
    const float* b_out  = (const float*)d_in[29];

    const int D = 128, H = 128, OUTN = 64;
    const int NC = in_sizes[0] / D;   // 100000
    const int NP = in_sizes[1] / D;   // 10000
    const int E  = in_sizes[2];       // 500000

    // workspace carve-out (peak ~65 MB)
    char* wp = (char*)d_ws;
    auto carve = [&](size_t bytes) {
        void* p = (void*)wp;
        wp += (bytes + 255) & ~(size_t)255;
        return p;
    };
    float*     h     = (float*)carve((size_t)NC * H * 4);   // h_gcn, then hs_gat
    float*     prod  = (float*)carve((size_t)NP * H * 4);
    float*     sum   = (float*)carve((size_t)NP * D * 4);
    float*     deg_s = (float*)carve((size_t)NC * 4);
    float*     deg_d = (float*)carve((size_t)NP * 4);
    float*     ls    = (float*)carve((size_t)NC * 4);
    float*     ldv   = (float*)carve((size_t)NP * 4);
    unsigned*  mb    = (unsigned*)carve((size_t)NP * 4);
    float*     den   = (float*)carve((size_t)NP * 4);
    float*     cnt   = (float*)carve((size_t)NP * 4);
    float*     elog  = (float*)carve((size_t)E * 4);
    float*     vs    = (float*)carve(128 * 4);
    float*     vd    = (float*)carve(128 * 4);
    float*     WrS   = (float*)carve(128 * 128 * 4);
    _Float16*  Bp    = (_Float16*)carve(128 * 128 * 2);     // packed weights (reused)

    auto cdiv = [](long a, long b) { return (unsigned)((a + b - 1) / b); };
    const unsigned packGrid8 = cdiv(4 * 8 * 32 * 16, TPB);  // N=128
    const unsigned packGrid4 = cdiv(4 * 4 * 32 * 16, TPB);  // N=64
    const unsigned gemmGridC = cdiv(NC / 16, 8);            // strips of 16 rows
    const unsigned gemmGridP = cdiv(NP / 16, 8);

    // ---- GCN (purchase) ----
    fill_f32<<<cdiv(NC, TPB), TPB, 0, stream>>>(deg_s, NC, 0.f);
    fill_f32<<<cdiv(NP, TPB), TPB, 0, stream>>>(deg_d, NP, 0.f);
    degree_count<<<cdiv(E, TPB), TPB, 0, stream>>>(src_pur, dst_pur, deg_s, deg_d, E);
    pack_b<8><<<packGrid8, TPB, 0, stream>>>(W_gcn, Bp);
    gemm_f16p<8, false, false, false><<<gemmGridC, TPB, 0, stream>>>(
        x_cust, Bp, h, NC, nullptr, nullptr);
    bias_init<<<cdiv((long)NP * H, TPB), TPB, 0, stream>>>(
        prod, b_gcn, b_gat, b_to, b_fr, b_dv, NP * H);
    scatter_gcn<<<cdiv((long)E * 32, TPB), TPB, 0, stream>>>(
        h, src_pur, dst_pur, deg_s, deg_d, prod, E);

    // ---- GAT (redeem): ls = x_cust @ (Ws@a_s), ld = x_prod @ (Wd@a_d) ----
    matvec128<<<1, TPB, 0, stream>>>(Ws_gat, a_s, vs);
    matvec128<<<1, TPB, 0, stream>>>(Wd_gat, a_d, vd);
    rowdot128<<<cdiv((long)NC * 32, TPB), TPB, 0, stream>>>(x_cust, vs, ls, NC);
    rowdot128<<<cdiv((long)NP * 32, TPB), TPB, 0, stream>>>(x_prod, vd, ldv, NP);
    fill_f32<<<cdiv(NP, TPB), TPB, 0, stream>>>((float*)mb, NP, 0.f); // enc(-inf)
    fill_f32<<<cdiv(NP, TPB), TPB, 0, stream>>>(den, NP, 0.f);
    gat_logit_max<<<cdiv(E, TPB), TPB, 0, stream>>>(src_red, dst_red, ls, ldv, elog, mb, E);
    gat_exp_den<<<cdiv(E, TPB), TPB, 0, stream>>>(dst_red, elog, mb, den, E);
    pack_b<8><<<packGrid8, TPB, 0, stream>>>(Ws_gat, Bp);  // hs = x_cust @ Ws_gat
    gemm_f16p<8, false, false, false><<<gemmGridC, TPB, 0, stream>>>(
        x_cust, Bp, h, NC, nullptr, nullptr);
    scatter_gat<<<cdiv((long)E * 32, TPB), TPB, 0, stream>>>(
        h, src_red, dst_red, elog, den, prod, E);

    // ---- SAGE x3 (neighbor half): prod += mean(x_cust[src]) @ Wl ----
    const int* s3[3]    = {src_to, src_fr, src_dv};
    const int* d3[3]    = {dst_to, dst_fr, dst_dv};
    const float* Wl3[3] = {Wl_to, Wl_fr, Wl_dv};
    for (int r = 0; r < 3; ++r) {
        fill_f32<<<cdiv((long)NP * D, TPB), TPB, 0, stream>>>(sum, NP * D, 0.f);
        fill_f32<<<cdiv(NP, TPB), TPB, 0, stream>>>(cnt, NP, 0.f);
        scatter_sage<<<cdiv((long)E * 32, TPB), TPB, 0, stream>>>(
            x_cust, s3[r], d3[r], sum, cnt, E);
        pack_b<8><<<packGrid8, TPB, 0, stream>>>(Wl3[r], Bp);
        gemm_f16p<8, false, true, true><<<gemmGridP, TPB, 0, stream>>>(
            sum, Bp, prod, NP, cnt, nullptr);
    }
    // ---- SAGE root halves fused: prod += x_prod @ (Wr_to+Wr_fr+Wr_dv) ----
    add3<<<cdiv(128 * 128, TPB), TPB, 0, stream>>>(Wr_to, Wr_fr, Wr_dv, WrS, 128 * 128);
    pack_b<8><<<packGrid8, TPB, 0, stream>>>(WrS, Bp);
    gemm_f16p<8, false, true, false><<<gemmGridP, TPB, 0, stream>>>(
        x_prod, Bp, prod, NP, nullptr, nullptr);

    // ---- out = relu(prod) @ W_out + b_out ----
    pack_b<4><<<packGrid4, TPB, 0, stream>>>(W_out, Bp);
    gemm_f16p<4, true, false, false><<<cdiv(NP / 16, 8), TPB, 0, stream>>>(
        prod, Bp, (float*)d_out, NP, nullptr, b_out);
}
